// NODE_32229434589744
// MI455X (gfx1250) — matrix operand order repaired and checked by reading, not compiled, not run on hardware
//
#include <hip/hip_runtime.h>
#include <hip/hip_bf16.h>
#include <cstdint>

#define BB 1024
#define TT 512
#define LL 512
#define DD 64
#define HH 128
#define IN 129      // 2*D+1
#define ING 160     // K padded to 5*32 for WMMA
#define G4 512      // 4*H

typedef __attribute__((ext_vector_type(16))) _Float16 v16h;
typedef __attribute__((ext_vector_type(8)))  float    v8f;

union AFrag { v16h v; _Float16 h[16]; uint4 q[2]; };
union CAcc  { v8f v; float f[8]; };

// A-fragment (16x32 f16, M x K) gathered from row-major f16 LDS.
// This file's contiguous k order, as in the B packing: lane (m = lane&15, hi = lane>>4) holds K in
// [k0+16*hi, k0+16*hi+16).
__device__ __forceinline__ v16h lds_a_frag(const _Float16* base, int stride, int k0, int lane) {
    int m = lane & 15, hi = lane >> 4;
    const _Float16* p = base + m * stride + k0 + 16 * hi;
    AFrag a;
    a.q[0] = *(const uint4*)(p);
    a.q[1] = *(const uint4*)(p + 8);
    return a.v;
}

__device__ __forceinline__ float sigm(float x) { return 1.0f / (1.0f + __expf(-x)); }

// ---------------------------------------------------------------------------
// Kernel 0: convert fp32 weights to f16, pre-swizzled to the WMMA B-fragment
// layout: tile[(lane*16)+j] = B(k = ks*32 + 16*(lane>>4) + j, n = nt*16 + lane&15)
// where B(k,n) = W[n][k] (GEMM is act @ W^T). W_ih K padded 129->160 with zeros.
// ---------------------------------------------------------------------------
#define F_IH (32*5*512)
#define F_HH (32*4*512)
#define F_D1 (8*2*512)
#define F_D2 (4*4*512)

__global__ void __launch_bounds__(256) convert_weights(
    const float* __restrict__ Wih, const float* __restrict__ Whh,
    const float* __restrict__ Wd1, const float* __restrict__ Wd2,
    _Float16* __restrict__ ws16)
{
    int idx = blockIdx.x * 256 + threadIdx.x;
    if (idx >= F_IH + F_HH + F_D1 + F_D2) return;
    float val;
    if (idx < F_IH) {
        int tile = idx >> 9, r = idx & 511;
        int lane = r >> 4, j = r & 15;
        int nt = tile / 5, ks = tile % 5;
        int n = nt * 16 + (lane & 15);
        int k = ks * 32 + ((lane >> 4) << 4) + j;
        val = (k < IN) ? Wih[n * IN + k] : 0.0f;
    } else if (idx < F_IH + F_HH) {
        int id = idx - F_IH;
        int tile = id >> 9, r = id & 511;
        int lane = r >> 4, j = r & 15;
        int nt = tile >> 2, ks = tile & 3;
        int n = nt * 16 + (lane & 15);
        int k = ks * 32 + ((lane >> 4) << 4) + j;
        val = Whh[n * HH + k];
    } else if (idx < F_IH + F_HH + F_D1) {
        int id = idx - F_IH - F_HH;
        int tile = id >> 9, r = id & 511;
        int lane = r >> 4, j = r & 15;
        int nt = tile >> 1, ks = tile & 1;
        int n = nt * 16 + (lane & 15);            // 0..127 (hidden)
        int k = ks * 32 + ((lane >> 4) << 4) + j; // 0..63  (D)
        val = Wd1[n * DD + k];
    } else {
        int id = idx - F_IH - F_HH - F_D1;
        int tile = id >> 9, r = id & 511;
        int lane = r >> 4, j = r & 15;
        int nt = tile >> 2, ks = tile & 3;
        int n = nt * 16 + (lane & 15);            // 0..63  (D)
        int k = ks * 32 + ((lane >> 4) << 4) + j; // 0..127 (hidden)
        val = Wd2[n * HH + k];
    }
    ws16[idx] = (_Float16)val;
}

// ---------------------------------------------------------------------------
// Kernel 1: fused LSTM. One block = 16 batch rows, 256 threads = 8 waves.
// All 36 weight B-fragments per wave live in registers for the whole t-loop
// (CDNA5 1024-VGPR file via VGPR-MSB addressing); x_t is double-buffered into
// LDS with global_load_async_to_lds_b32 (ASYNCcnt) so step t+1's HBM latency
// hides under step t's WMMAs. Steady-state step touches only LDS + VALU.
// ---------------------------------------------------------------------------
__global__ void __launch_bounds__(256) lstm_kernel(
    const float* __restrict__ x, const int* __restrict__ input_len,
    const float* __restrict__ b_ih, const float* __restrict__ b_hh,
    const _Float16* __restrict__ wih, const _Float16* __restrict__ whh,
    float* __restrict__ pool_out)
{
    __shared__ __align__(16) _Float16 lx[16 * ING];
    __shared__ __align__(16) _Float16 lh[16 * HH];
    __shared__ __align__(16) float    lg[16 * G4];
    __shared__ __align__(16) float    xstage[2][16 * IN];

    int tid  = threadIdx.x;
    int lane = tid & 31, w = tid >> 5;
    int b0   = blockIdx.x * 16;

    for (int i = tid; i < 16 * ING; i += 256) lx[i] = (_Float16)0.0f;
    for (int i = tid; i < 16 * HH;  i += 256) lh[i] = (_Float16)0.0f;

    // Pointwise ownership: row mrow, 8 consecutive H-columns from j0.
    int mrow = tid >> 4;
    int j0   = (tid & 15) * 8;
    float c[8], pool[8];
#pragma unroll
    for (int q = 0; q < 8; ++q) { c[q] = 0.0f; pool[q] = 0.0f; }
    int len = input_len[b0 + mrow];

    // Pin this wave's weight fragments in registers: 4 N-tiles x (5 ih + 4 hh).
    v16h wf[36];
    float biasq[4];
#pragma unroll
    for (int q = 0; q < 4; ++q) {
        int nt = w * 4 + q;
#pragma unroll
        for (int ks = 0; ks < 5; ++ks)
            wf[q * 9 + ks] = *(const v16h*)(wih + (size_t)(nt * 5 + ks) * 512 + lane * 16);
#pragma unroll
        for (int ks = 0; ks < 4; ++ks)
            wf[q * 9 + 5 + ks] = *(const v16h*)(whh + (size_t)(nt * 4 + ks) * 512 + lane * 16);
        int n = nt * 16 + (lane & 15);
        biasq[q] = b_ih[n] + b_hh[n];
    }

    // async copy of x_t (16 x 129 f32) into staging buffer `buf`
    auto issue_async = [&](int t, int buf) {
        for (int i = tid; i < 16 * IN; i += 256) {
            int m = i / IN, k = i - m * IN;
            unsigned lds_addr = (unsigned)(uintptr_t)&xstage[buf][i];
            const float* g = &x[((size_t)(b0 + m) * TT + t) * IN + k];
            asm volatile("global_load_async_to_lds_b32 %0, %1, off"
                         :: "v"(lds_addr), "v"(g) : "memory");
        }
    };

    issue_async(0, 0);
    __syncthreads();

    for (int t = 0; t < TT; ++t) {
        // drain this step's async tile; barrier makes all waves' LDS visible
        asm volatile("s_wait_asynccnt 0x0" ::: "memory");
        __syncthreads();

        // convert staged f32 -> f16 WMMA staging (pad cols stay 0)
        int cur = t & 1;
        for (int i = tid; i < 16 * IN; i += 256) {
            int m = i / IN, k = i - m * IN;
            lx[m * ING + k] = (_Float16)xstage[cur][i];
        }
        // prefetch next step while this step computes
        if (t + 1 < TT) issue_async(t + 1, (t + 1) & 1);
        __syncthreads();

        // A-fragments shared by all four N-tiles of this wave: hoist them
        v16h ax[5], ah[4];
#pragma unroll
        for (int ks = 0; ks < 5; ++ks) ax[ks] = lds_a_frag(lx, ING, ks * 32, lane);
#pragma unroll
        for (int ks = 0; ks < 4; ++ks) ah[ks] = lds_a_frag(lh, HH, ks * 32, lane);

        // gates = x_t @ W_ih^T + h @ W_hh^T + bias  (16 x 512)
#pragma unroll
        for (int q = 0; q < 4; ++q) {
            int nt = w * 4 + q;
            CAcc acc;
#pragma unroll
            for (int r = 0; r < 8; ++r) acc.f[r] = 0.0f;
#pragma unroll
            for (int ks = 0; ks < 5; ++ks)
                acc.v = __builtin_amdgcn_wmma_f32_16x16x32_f16(
                    false, ax[ks], false, wf[q * 9 + ks], (short)0, acc.v, false, false);
#pragma unroll
            for (int ks = 0; ks < 4; ++ks)
                acc.v = __builtin_amdgcn_wmma_f32_16x16x32_f16(
                    false, ah[ks], false, wf[q * 9 + 5 + ks], (short)0, acc.v, false, false);
            int n  = nt * 16 + (lane & 15);
            int hi = lane >> 4;
#pragma unroll
            for (int r = 0; r < 8; ++r) {
                int m = r + 8 * hi;
                lg[m * G4 + n] = acc.f[r] + biasq[q];
            }
        }
        __syncthreads();

        // pointwise LSTM cell + masked pooling; write h back as f16
#pragma unroll
        for (int q = 0; q < 8; ++q) {
            int j = j0 + q;
            float gi = lg[mrow * G4 + j];
            float gf = lg[mrow * G4 + HH + j];
            float gg = lg[mrow * G4 + 2 * HH + j];
            float go = lg[mrow * G4 + 3 * HH + j];
            c[q] = sigm(gf) * c[q] + sigm(gi) * tanhf(gg);
            float h = sigm(go) * tanhf(c[q]);
            if (t < len) pool[q] += h;
            lh[mrow * HH + j] = (_Float16)h;
        }
        __syncthreads();
    }

    float inv = 1.0f / (float)len;
#pragma unroll
    for (int q = 0; q < 8; ++q)
        pool_out[(size_t)(b0 + mrow) * HH + j0 + q] = pool[q] * inv;
}

// ---------------------------------------------------------------------------
// Kernel 2: init = relu(pool@Wp1^T+bp1)@Wp2^T+bp2 -> pred[:,0,:]  (tiny)
// ---------------------------------------------------------------------------
__global__ void __launch_bounds__(128) proj_kernel(
    const float* __restrict__ pool,
    const float* __restrict__ Wp1, const float* __restrict__ bp1,
    const float* __restrict__ Wp2, const float* __restrict__ bp2,
    float* __restrict__ pred)
{
    __shared__ float hid[HH];
    int b = blockIdx.x, j = threadIdx.x;
    const float* p = pool + (size_t)b * HH;
    float s = bp1[j];
    for (int k = 0; k < HH; ++k) s += p[k] * Wp1[j * HH + k];
    hid[j] = fmaxf(s, 0.0f);
    __syncthreads();
    if (j < DD) {
        float o = bp2[j];
        for (int k = 0; k < HH; ++k) o += hid[k] * Wp2[j * HH + k];
        pred[(size_t)b * LL * DD + j] = o;
    }
}

// ---------------------------------------------------------------------------
// Kernel 3: RK4 Neural-ODE. One block = 16 batch rows, 128 threads = 4 waves.
// f(y) = relu(y@Wd1^T+bd1)@Wd2^T+bd2; weight fragments register-resident for
// the whole 511-step integration; all GEMMs via WMMA, state in LDS.
// ---------------------------------------------------------------------------
__global__ void __launch_bounds__(128) rk4_kernel(
    const float* __restrict__ ltime,
    const _Float16* __restrict__ wd1, const _Float16* __restrict__ wd2,
    const float* __restrict__ bd1, const float* __restrict__ bd2,
    float* __restrict__ pred)
{
    __shared__ __align__(16) float    ly[16 * DD];
    __shared__ __align__(16) float    lkin[16 * DD];
    __shared__ __align__(16) float    lacc[16 * DD];
    __shared__ __align__(16) _Float16 lyt[16 * DD];
    __shared__ __align__(16) _Float16 lhid[16 * HH];
    __shared__ float ldt[16];

    int tid = threadIdx.x, lane = tid & 31, w = tid >> 5;
    int b0  = blockIdx.x * 16;
    int mrow = tid >> 3;          // 8 threads per row
    int d0   = (tid & 7) * 8;

    // Pin weight fragments: stage-1 2 N-tiles x 2 ks, stage-2 1 N-tile x 4 ks.
    v16h wf1[2][2], wf2[4];
    float bb1[2], bb2;
#pragma unroll
    for (int q = 0; q < 2; ++q) {
        int nt = w * 2 + q;
#pragma unroll
        for (int ks = 0; ks < 2; ++ks)
            wf1[q][ks] = *(const v16h*)(wd1 + (size_t)(nt * 2 + ks) * 512 + lane * 16);
        bb1[q] = bd1[nt * 16 + (lane & 15)];
    }
#pragma unroll
    for (int ks = 0; ks < 4; ++ks)
        wf2[ks] = *(const v16h*)(wd2 + (size_t)(w * 4 + ks) * 512 + lane * 16);
    bb2 = bd2[w * 16 + (lane & 15)];

#pragma unroll
    for (int q = 0; q < 8; ++q)
        ly[mrow * DD + d0 + q] = pred[(size_t)(b0 + mrow) * LL * DD + d0 + q];
    __syncthreads();

    const float wgt[4] = {1.0f, 2.0f, 2.0f, 1.0f};
    const float alp[4] = {0.0f, 0.5f, 0.5f, 1.0f};

    for (int l = 1; l < LL; ++l) {
        if (tid < 16)
            ldt[tid] = ltime[(size_t)(b0 + tid) * LL + l] -
                       ltime[(size_t)(b0 + tid) * LL + l - 1];
#pragma unroll
        for (int q = 0; q < 8; ++q) lacc[mrow * DD + d0 + q] = 0.0f;
        __syncthreads();

        for (int s = 0; s < 4; ++s) {
            // ytmp = y + alpha*dt*k_prev  (f16)
            float a = alp[s], dtm = ldt[mrow];
#pragma unroll
            for (int q = 0; q < 8; ++q) {
                float yv = ly[mrow * DD + d0 + q];
                float kv = (s == 0) ? 0.0f : lkin[mrow * DD + d0 + q];
                lyt[mrow * DD + d0 + q] = (_Float16)(yv + a * dtm * kv);
            }
            __syncthreads();

            // hidden = relu(ytmp @ Wd1^T + bd1): 8 N-tiles (2 per wave), K=64
            v16h ay[2];
#pragma unroll
            for (int ks = 0; ks < 2; ++ks) ay[ks] = lds_a_frag(lyt, DD, ks * 32, lane);
#pragma unroll
            for (int q = 0; q < 2; ++q) {
                int nt = w * 2 + q;
                CAcc acc;
#pragma unroll
                for (int r = 0; r < 8; ++r) acc.f[r] = 0.0f;
#pragma unroll
                for (int ks = 0; ks < 2; ++ks)
                    acc.v = __builtin_amdgcn_wmma_f32_16x16x32_f16(
                        false, ay[ks], false, wf1[q][ks], (short)0, acc.v, false, false);
                int n = nt * 16 + (lane & 15), hi = lane >> 4;
#pragma unroll
                for (int r = 0; r < 8; ++r)
                    lhid[(r + 8 * hi) * HH + n] = (_Float16)fmaxf(acc.f[r] + bb1[q], 0.0f);
            }
            __syncthreads();

            // k = hidden @ Wd2^T + bd2: 4 N-tiles (1 per wave), K=128
            {
                CAcc acc;
#pragma unroll
                for (int r = 0; r < 8; ++r) acc.f[r] = 0.0f;
#pragma unroll
                for (int ks = 0; ks < 4; ++ks) {
                    v16h av = lds_a_frag(lhid, HH, ks * 32, lane);
                    acc.v = __builtin_amdgcn_wmma_f32_16x16x32_f16(
                        false, av, false, wf2[ks], (short)0, acc.v, false, false);
                }
                int n = w * 16 + (lane & 15), hi = lane >> 4;
                float wg = wgt[s];
#pragma unroll
                for (int r = 0; r < 8; ++r) {
                    int m = r + 8 * hi;
                    float kv = acc.f[r] + bb2;
                    lkin[m * DD + n]  = kv;
                    lacc[m * DD + n] += wg * kv;
                }
            }
            __syncthreads();
        }

        // y += dt/6 * (k1+2k2+2k3+k4); stream to pred
        float dtm = ldt[mrow];
#pragma unroll
        for (int q = 0; q < 8; ++q) {
            float y2 = ly[mrow * DD + d0 + q] + dtm * (1.0f / 6.0f) * lacc[mrow * DD + d0 + q];
            ly[mrow * DD + d0 + q] = y2;
            pred[((size_t)(b0 + mrow) * LL + l) * DD + d0 + q] = y2;
        }
        __syncthreads();
    }
}

// ---------------------------------------------------------------------------
// Kernel 4: per-sample masked-MSE partials (deterministic block reduction)
// ---------------------------------------------------------------------------
__global__ void __launch_bounds__(256) loss_partial(
    const float* __restrict__ pred, const float* __restrict__ lfeat,
    const int* __restrict__ lmask, const int* __restrict__ stype,
    float* __restrict__ part)
{
    __shared__ float ra[256], rb[256], rc[256];
    __shared__ int   ia[256], ib[256];
    int b = blockIdx.x, tid = threadIdx.x;
    float s_all = 0.0f, s1 = 0.0f, s2 = 0.0f;
    for (int idx = tid; idx < LL * DD; idx += 256) {
        int l = idx >> 6;
        size_t gi = (size_t)b * LL * DD + idx;
        float e = pred[gi] - lfeat[gi];
        float err = e * e * (1.0f - (float)lmask[gi]);
        s_all += err;
        int st = stype[b * LL + l];
        if (st == 1) s1 += err; else if (st == 2) s2 += err;
    }
    int c1 = 0, c2 = 0;
    for (int l = tid; l < LL; l += 256) {
        int st = stype[b * LL + l];
        c1 += (st == 1); c2 += (st == 2);
    }
    ra[tid] = s_all; rb[tid] = s1; rc[tid] = s2; ia[tid] = c1; ib[tid] = c2;
    __syncthreads();
    for (int off = 128; off > 0; off >>= 1) {
        if (tid < off) {
            ra[tid] += ra[tid + off]; rb[tid] += rb[tid + off]; rc[tid] += rc[tid + off];
            ia[tid] += ia[tid + off]; ib[tid] += ib[tid + off];
        }
        __syncthreads();
    }
    if (tid == 0) {
        float rl = (float)ia[0], pl = (float)ib[0];
        part[b * 3 + 0] = ra[0] / (rl + pl) / (float)DD;
        part[b * 3 + 1] = rb[0] / rl / (float)DD;
        part[b * 3 + 2] = rc[0] / pl / (float)DD;
    }
}

__global__ void __launch_bounds__(256) loss_final(const float* __restrict__ part,
                                                  float* __restrict__ out3)
{
    __shared__ float r0[256], r1[256], r2[256];
    int tid = threadIdx.x;
    float a = 0.0f, b = 0.0f, c = 0.0f;
    for (int i = tid; i < BB; i += 256) {
        a += part[i * 3 + 0]; b += part[i * 3 + 1]; c += part[i * 3 + 2];
    }
    r0[tid] = a; r1[tid] = b; r2[tid] = c;
    __syncthreads();
    for (int off = 128; off > 0; off >>= 1) {
        if (tid < off) { r0[tid] += r0[tid + off]; r1[tid] += r1[tid + off]; r2[tid] += r2[tid + off]; }
        __syncthreads();
    }
    if (tid == 0) {
        out3[0] = r0[0] / (float)BB;
        out3[1] = r1[0] / (float)BB;
        out3[2] = r2[0] / (float)BB;
    }
}

// ---------------------------------------------------------------------------
extern "C" void kernel_launch(void* const* d_in, const int* in_sizes, int n_in,
                              void* d_out, int out_size, void* d_ws, size_t ws_size,
                              hipStream_t stream) {
    const float* x     = (const float*)d_in[0];
    const float* ltime = (const float*)d_in[1];
    const float* lfeat = (const float*)d_in[2];
    const int*   lmask = (const int*)d_in[3];
    const int*   stype = (const int*)d_in[4];
    const int*   ilen  = (const int*)d_in[5];
    const float* Wih   = (const float*)d_in[6];
    const float* Whh   = (const float*)d_in[7];
    const float* bih   = (const float*)d_in[8];
    const float* bhh   = (const float*)d_in[9];
    const float* Wp1   = (const float*)d_in[10];
    const float* bp1   = (const float*)d_in[11];
    const float* Wp2   = (const float*)d_in[12];
    const float* bp2   = (const float*)d_in[13];
    const float* Wd1   = (const float*)d_in[14];
    const float* bd1   = (const float*)d_in[15];
    const float* Wd2   = (const float*)d_in[16];
    const float* bd2   = (const float*)d_in[17];

    float* pred = (float*)d_out;
    float* out3 = pred + (size_t)BB * LL * DD;

    _Float16* ws16  = (_Float16*)d_ws;
    _Float16* wih16 = ws16;
    _Float16* whh16 = wih16 + F_IH;
    _Float16* wd116 = whh16 + F_HH;
    _Float16* wd216 = wd116 + F_D1;
    float* pool = (float*)(wd216 + F_D2);
    float* part = pool + (size_t)BB * HH;

    int totalw = F_IH + F_HH + F_D1 + F_D2;
    convert_weights<<<(totalw + 255) / 256, 256, 0, stream>>>(Wih, Whh, Wd1, Wd2, ws16);
    lstm_kernel<<<BB / 16, 256, 0, stream>>>(x, ilen, bih, bhh, wih16, whh16, pool);
    proj_kernel<<<BB, 128, 0, stream>>>(pool, Wp1, bp1, Wp2, bp2, pred);
    rk4_kernel<<<BB / 16, 128, 0, stream>>>(ltime, wd116, wd216, bd1, bd2, pred);
    loss_partial<<<BB, 256, 0, stream>>>(pred, lfeat, lmask, stype, part);
    loss_final<<<1, 256, 0, stream>>>(part, out3);
}
